// Self_Attention_7224134992096
// MI455X (gfx1250) — matrix operand order repaired and checked by reading, not compiled, not run on hardware
//
#include <hip/hip_runtime.h>
#include <hip/hip_bf16.h>

// ---------------------------------------------------------------------------
// Problem constants
// ---------------------------------------------------------------------------
#define BB 32
#define CC 256
#define WW 32
#define HH 32
#define NN 1024   // W*H

typedef __bf16 bf16;
typedef __attribute__((ext_vector_type(8)))  __bf16 v8bf;
typedef __attribute__((ext_vector_type(16))) __bf16 v16bf;
typedef __attribute__((ext_vector_type(8)))  float  v8f;

// ---------------------------------------------------------------------------
// Fragment loaders (wave32 WMMA layouts, bf16 16x16x32)
//   A (16xK=32): lane holds row m=lane&15; K chunks at kbase+8*half and +16
//   B (K=32x16): lane holds col n=lane&15; K chunks at kbase+8*half and +16
// ---------------------------------------------------------------------------
__device__ __forceinline__ v16bf ldA(const bf16* p) {
  v8bf lo = *(const v8bf*)p;
  v8bf hi = *(const v8bf*)(p + 16);
  return __builtin_shufflevector(lo, hi, 0,1,2,3,4,5,6,7,8,9,10,11,12,13,14,15);
}
__device__ __forceinline__ v16bf ldB(const bf16* p) {
  v8bf lo = *(const v8bf*)p;
  v8bf hi = *(const v8bf*)(p + 16);
  return __builtin_shufflevector(lo, hi, 0,1,2,3,4,5,6,7,8,9,10,11,12,13,14,15);
}
__device__ __forceinline__ v8f wmma_bf16(v16bf a, v16bf b, v8f c) {
  return __builtin_amdgcn_wmma_f32_16x16x32_bf16(
      /*neg_a=*/false, a, /*neg_b=*/false, b,
      /*c_mod=*/(short)0, c, /*reuse_a=*/false, /*reuse_b=*/false);
}

// ---------------------------------------------------------------------------
// Prep kernels
// ---------------------------------------------------------------------------
// xT[b][n][c] = (bf16) x[b][c][n]
__global__ void xpose_kernel(const float* __restrict__ x, bf16* __restrict__ xT) {
  unsigned tid = blockIdx.x * blockDim.x + threadIdx.x;
  unsigned c = tid & (CC - 1);
  unsigned n = (tid >> 8) & (NN - 1);
  unsigned b = tid >> 18;
  xT[((size_t)b * NN + n) * CC + c] = (bf16)x[((size_t)b * CC + c) * NN + n];
}

// elementwise fp32 -> bf16
__global__ void cvt_kernel(const float* __restrict__ s, bf16* __restrict__ d, int n) {
  int i = blockIdx.x * blockDim.x + threadIdx.x;
  if (i < n) d[i] = (bf16)s[i];
}

// relT[n][c] = rel_h[c][h] + rel_w[c][w],  n = w*H + h
__global__ void rel_kernel(const float* __restrict__ rh, const float* __restrict__ rw,
                           bf16* __restrict__ relT) {
  unsigned tid = blockIdx.x * blockDim.x + threadIdx.x;
  unsigned c = tid & (CC - 1);
  unsigned n = tid >> 8;
  unsigned h = n & (HH - 1);
  unsigned w = n >> 5;
  relT[(size_t)n * CC + c] = (bf16)(rh[c * HH + h] + rw[c * WW + w]);
}

// ---------------------------------------------------------------------------
// QKV projection: one wave per 16(o) x 16(n) tile, all three projections.
// qT/kT stored [B][N][C] bf16 (transposed); V stored [B][C][N] bf16.
// grid = (C/16, N/16, B), block = 32
// ---------------------------------------------------------------------------
__global__ __launch_bounds__(32)
void qkv_kernel(const bf16* __restrict__ xT,
                const bf16* __restrict__ Wq, const bf16* __restrict__ Wk,
                const bf16* __restrict__ Wv,
                const float* __restrict__ bq, const float* __restrict__ bk,
                const float* __restrict__ bv,
                bf16* __restrict__ qT, bf16* __restrict__ kT,
                bf16* __restrict__ vv) {
  const int ot = blockIdx.x, nt = blockIdx.y, b = blockIdx.z;
  const int lane = threadIdx.x;
  const int half = lane >> 4, lr = lane & 15;

  v8f accq = {}, acck = {}, accv = {};
  const int ncol = nt * 16 + lr;
  const bf16* xcolBase = xT + ((size_t)b * NN + ncol) * CC;

#pragma unroll
  for (int kb = 0; kb < 8; ++kb) {
    const int kbase = kb * 32;
    v16bf bx = ldB(xcolBase + kbase + half * 8);
    v16bf aq = ldA(Wq + (size_t)(ot * 16 + lr) * CC + kbase + half * 8);
    v16bf ak = ldA(Wk + (size_t)(ot * 16 + lr) * CC + kbase + half * 8);
    v16bf av = ldA(Wv + (size_t)(ot * 16 + lr) * CC + kbase + half * 8);
    accq = wmma_bf16(aq, bx, accq);
    acck = wmma_bf16(ak, bx, acck);
    accv = wmma_bf16(av, bx, accv);
  }

  // D layout: VGPR r -> row m = r + 8*half, col n = lr
  v8bf qb, kb_;
#pragma unroll
  for (int r = 0; r < 8; ++r) {
    const int oc = ot * 16 + half * 8 + r;
    qb[r]  = (bf16)(accq[r] + bq[oc]);
    kb_[r] = (bf16)(acck[r] + bk[oc]);
    vv[((size_t)b * CC + oc) * NN + ncol] = (bf16)(accv[r] + bv[oc]);
  }
  *(v8bf*)(qT + ((size_t)b * NN + ncol) * CC + ot * 16 + half * 8) = qb;
  *(v8bf*)(kT + ((size_t)b * NN + ncol) * CC + ot * 16 + half * 8) = kb_;
}

// ---------------------------------------------------------------------------
// Attention: 4 waves (128 thr) per (b, 16-row tile). Logits tile (16 x 1024
// f32, padded) in dynamic LDS; waves split the j-dim for logits, rows for
// softmax, c-tiles for P*V. After softmax P is stored once as bf16 (padded)
// so PV B-fragments are raw ds_load_b128s.
//   L[i,j] = sum_c q[c,i]*k[c,j] + sum_c rel[c,i]*q[c,j]
//   out[c,i] = sum_j v[c,j] * softmax(L)[i,j]
// grid = (N/16, B), block = 128, dyn-LDS = 16*(NN+4)*4 + 16*(NN+8)*2
// ---------------------------------------------------------------------------
#define PSTR  (NN + 4)   // f32 row stride (bank-conflict padding)
#define PBSTR (NN + 8)   // bf16 row stride
#define ATTN_LDS (16 * PSTR * 4 + 16 * PBSTR * 2)

__global__ __launch_bounds__(128)
void attn_kernel(const bf16* __restrict__ qT, const bf16* __restrict__ kT,
                 const bf16* __restrict__ vv, const bf16* __restrict__ relT,
                 float* __restrict__ out) {
  extern __shared__ char smem[];
  float* P  = (float*)smem;                       // [16][PSTR]
  bf16*  Pb = (bf16*)(smem + 16 * PSTR * 4);      // [16][PBSTR]

  const int it = blockIdx.x, b = blockIdx.y;
  const int tid = threadIdx.x;
  const int wave = tid >> 5, lane = tid & 31;
  const int half = (lane >> 4) & 1, lr = lane & 15;
  const int irow = it * 16 + lr;

  // Hoisted A fragments: q rows (content-content) and rel rows (content-pos)
  v16bf qA[8], rA[8];
#pragma unroll
  for (int kb = 0; kb < 8; ++kb) {
    qA[kb] = ldA(qT + ((size_t)b * NN + irow) * CC + kb * 32 + half * 8);
    rA[kb] = ldA(relT + (size_t)irow * CC + kb * 32 + half * 8);
  }

  // ---- logits: each wave covers 16 j-tiles, processed in pairs ----
  for (int jp = 0; jp < 8; ++jp) {
    const int jt0 = wave * 16 + jp * 2;
    const int jc0 = jt0 * 16 + lr;
    const bf16* k0 = kT + ((size_t)b * NN + jc0) * CC;
    const bf16* k1 = k0 + 16 * CC;
    const bf16* q0 = qT + ((size_t)b * NN + jc0) * CC;
    const bf16* q1 = q0 + 16 * CC;
    __builtin_prefetch(k1 + 16 * CC, 0, 1);

    v8f a0 = {}, a1 = {};
#pragma unroll
    for (int kb = 0; kb < 8; ++kb) {
      const int o = kb * 32 + half * 8;
      a0 = wmma_bf16(qA[kb], ldB(k0 + o), a0);
      a1 = wmma_bf16(qA[kb], ldB(k1 + o), a1);
    }
#pragma unroll
    for (int kb = 0; kb < 8; ++kb) {
      const int o = kb * 32 + half * 8;
      a0 = wmma_bf16(rA[kb], ldB(q0 + o), a0);
      a1 = wmma_bf16(rA[kb], ldB(q1 + o), a1);
    }
#pragma unroll
    for (int r = 0; r < 8; ++r) {
      P[(half * 8 + r) * PSTR + jt0 * 16 + lr]      = a0[r];
      P[(half * 8 + r) * PSTR + jt0 * 16 + 16 + lr] = a1[r];
    }
  }
  __syncthreads();

  // ---- softmax over j: wave w owns rows 4w..4w+3, 32 lanes per row ----
  for (int r = wave * 4; r < wave * 4 + 4; ++r) {
    float mx = -3.402823466e38f;
    for (int j = lane; j < NN; j += 32) mx = fmaxf(mx, P[r * PSTR + j]);
#pragma unroll
    for (int off = 16; off > 0; off >>= 1)
      mx = fmaxf(mx, __shfl_xor(mx, off, 32));
    float sum = 0.f;
    for (int j = lane; j < NN; j += 32) {
      float e = __expf(P[r * PSTR + j] - mx);
      P[r * PSTR + j] = e;
      sum += e;
    }
#pragma unroll
    for (int off = 16; off > 0; off >>= 1)
      sum += __shfl_xor(sum, off, 32);
    const float inv = 1.0f / sum;
    for (int j = lane; j < NN; j += 32)
      Pb[r * PBSTR + j] = (bf16)(P[r * PSTR + j] * inv);
  }
  __syncthreads();

  // ---- P*V: wave w owns c-tiles 4w..4w+3, pairs share the B fragment ----
  for (int cp = 0; cp < 2; ++cp) {
    const int ct0 = wave * 4 + cp * 2;
    const bf16* v0 = vv + ((size_t)b * CC + ct0 * 16 + lr) * NN;
    const bf16* v1 = v0 + 16 * NN;
    v8f a0 = {}, a1 = {};
    for (int js = 0; js < NN / 32; ++js) {
      const int jb = js * 32;
      // B = P^T: lane = col i=lr, K=j chunks at jb+8*half and +16 from bf16 LDS row lr
      v16bf pB = ldB(Pb + lr * PBSTR + jb + 8 * half);
      a0 = wmma_bf16(ldA(v0 + jb + half * 8), pB, a0);
      a1 = wmma_bf16(ldA(v1 + jb + half * 8), pB, a1);
    }
#pragma unroll
    for (int r = 0; r < 8; ++r) {
      const int c0 = ct0 * 16 + half * 8 + r;
      out[((size_t)b * CC + c0) * NN + it * 16 + lr]        = a0[r];
      out[((size_t)b * CC + c0 + 16) * NN + it * 16 + lr]   = a1[r];
    }
  }
}

// ---------------------------------------------------------------------------
// Launch: inputs (x, Wq, bq, Wk, bk, Wv, bv, rel_h, rel_w), out f32 [B,C,W,H]
// ---------------------------------------------------------------------------
extern "C" void kernel_launch(void* const* d_in, const int* in_sizes, int n_in,
                              void* d_out, int out_size, void* d_ws, size_t ws_size,
                              hipStream_t stream) {
  const float* x  = (const float*)d_in[0];
  const float* Wq = (const float*)d_in[1];
  const float* bq = (const float*)d_in[2];
  const float* Wk = (const float*)d_in[3];
  const float* bk = (const float*)d_in[4];
  const float* Wv = (const float*)d_in[5];
  const float* bv = (const float*)d_in[6];
  const float* rh = (const float*)d_in[7];
  const float* rw = (const float*)d_in[8];
  float* out = (float*)d_out;

  char* ws = (char*)d_ws;
  const size_t MB = 1u << 20;
  bf16* xT   = (bf16*)(ws);                       // 16 MB  [B][N][C]
  bf16* qT   = (bf16*)(ws + 16 * MB);             // 16 MB  [B][N][C]
  bf16* kT   = (bf16*)(ws + 32 * MB);             // 16 MB  [B][N][C]
  bf16* vv   = (bf16*)(ws + 48 * MB);             // 16 MB  [B][C][N]
  bf16* relT = (bf16*)(ws + 64 * MB);             // 512 KB [N][C]
  bf16* wqb  = (bf16*)(ws + 64 * MB + 512 * 1024);          // 128 KB
  bf16* wkb  = (bf16*)(ws + 64 * MB + 512 * 1024 + 128 * 1024);
  bf16* wvb  = (bf16*)(ws + 64 * MB + 512 * 1024 + 256 * 1024);

  // prep
  xpose_kernel<<<(BB * CC * NN) / 256, 256, 0, stream>>>(x, xT);
  cvt_kernel<<<(CC * CC) / 256, 256, 0, stream>>>(Wq, wqb, CC * CC);
  cvt_kernel<<<(CC * CC) / 256, 256, 0, stream>>>(Wk, wkb, CC * CC);
  cvt_kernel<<<(CC * CC) / 256, 256, 0, stream>>>(Wv, wvb, CC * CC);
  rel_kernel<<<(NN * CC) / 256, 256, 0, stream>>>(rh, rw, relT);

  // QKV projection
  qkv_kernel<<<dim3(CC / 16, NN / 16, BB), 32, 0, stream>>>(
      xT, wqb, wkb, wvb, bq, bk, bv, qT, kT, vv);

  // fused logits + softmax + PV (4 waves/block, ~97KB dynamic LDS)
  attn_kernel<<<dim3(NN / 16, BB), 128, ATTN_LDS, stream>>>(qT, kT, vv, relT, out);
}
